// UE2_joint_precoding_62113817035448
// MI455X (gfx1250) — matrix operand order repaired and checked by reading, not compiled, not run on hardware
//
#include <hip/hip_runtime.h>
#include <hip/hip_bf16.h>
#include <stdint.h>

// ---------------------------------------------------------------------------
// Problem constants (from reference)
// ---------------------------------------------------------------------------
static constexpr int BATCH = 4096;
static constexpr int TX    = 32;
static constexpr int SUB   = 13;
static constexpr int ROWS  = BATCH * SUB;   // 53248 rows per tower
static constexpr float EPS = 1e-3f;

// WMMA vector types (wave32, gfx1250)
typedef _Float16 half8  __attribute__((ext_vector_type(8)));
typedef _Float16 half16 __attribute__((ext_vector_type(16)));
typedef float    float8 __attribute__((ext_vector_type(8)));

union AFrag { half16 v; half8 h[2]; };

// LDS activation row stride in halves (1032*2 = 2064 B: 16B-aligned, rotates
// rows by 4 banks to break systematic bank conflicts)
static constexpr int STR = 1032;
static constexpr int MT  = 64;              // rows per block (M tile)

// ---------------------------------------------------------------------------
// Kernel 1: input normalization  ->  x[t][row][0..63] (f16)
// norm over (tx, re/im) per (b, sub, ue);  x col i: ue = i>>5, tx = i&31
// ---------------------------------------------------------------------------
__global__ void k_prep(const float* __restrict__ V1, const float* __restrict__ V2,
                       _Float16* __restrict__ x) {
    const int row = blockIdx.x;             // b*SUB + sub
    const int b   = row / SUB;
    const int sub = row - b * SUB;
    const int i   = threadIdx.x;            // 0..63
    const int ue  = i >> 5;
    const int tx  = i & 31;
    const float* V = ue ? V2 : V1;
    const size_t base = (((size_t)b * TX + tx) * SUB + sub) * 2;
    const float a0 = V[base + 0];
    const float a1 = V[base + 1];
    float s = a0 * a0 + a1 * a1;            // reduce over the 32 lanes (one ue / wave)
    #pragma unroll
    for (int m = 16; m; m >>= 1) s += __shfl_xor(s, m, 32);
    const float rinv = rsqrtf(s);
    x[((size_t)0 * ROWS + row) * 64 + i] = (_Float16)(a0 * rinv);
    x[((size_t)1 * ROWS + row) * 64 + i] = (_Float16)(a1 * rinv);
}

// ---------------------------------------------------------------------------
// Kernel 2: repack weights (2,K,N) f32 -> f16 WMMA B-fragment order.
// Fragment (kt,nt): 32 lanes x 16 halves; lane l: n = nt*16 + (l&15),
// halves e: k = kt*32 + 8*(l>>4) + (e&7) + 16*(e>>3).   dst flat index == idx by design.
// ---------------------------------------------------------------------------
template <int K, int N>
__global__ void k_repack(const float* __restrict__ W, _Float16* __restrict__ dst) {
    const int idx = blockIdx.x * 256 + threadIdx.x;
    if (idx >= 2 * K * N) return;
    const int t    = idx / (K * N);
    const int r    = idx - t * (K * N);
    const int frag = r >> 9;                // 512 halves per fragment
    const int lane = (r >> 4) & 31;
    const int e    = r & 15;
    const int nt   = frag % (N / 16);
    const int kt   = frag / (N / 16);
    const int n    = nt * 16 + (lane & 15);
    const int k    = kt * 32 + ((lane >> 4) << 3) + (e & 7) + ((e >> 3) << 4);
    dst[idx] = (_Float16)W[(size_t)t * K * N + (size_t)k * N + n];
}

// ---------------------------------------------------------------------------
// Kernel 2b: fold BN params into (scale, shift): out = relu(h)*sc + sh
// ---------------------------------------------------------------------------
template <int N>
__global__ void k_bnprep(const float* __restrict__ g, const float* __restrict__ be,
                         const float* __restrict__ m, const float* __restrict__ v,
                         float* __restrict__ sc, float* __restrict__ sh) {
    const int idx = blockIdx.x * 256 + threadIdx.x;
    if (idx >= 2 * N) return;
    const float s = g[idx] * rsqrtf(v[idx] + EPS);
    sc[idx] = s;
    sh[idx] = be[idx] - m[idx] * s;
}

// ---------------------------------------------------------------------------
// Fused MLP layer: D = BN(ReLU(A*W + b)) (or +b only for the head).
// A from LDS (f16, padded rows), W from pre-fragmented global (L2-resident).
// Wave grid: wm in {0..3} (16-row strips), wn in {0,1} (N/2-column strips).
// G n-tiles share one A fragment -> 4 (or 2) independent WMMA chains.
// ---------------------------------------------------------------------------
template <int K, int N, bool BN>
__device__ __forceinline__ void layer(const _Float16* __restrict__ wf,
                                      const float* __restrict__ bias,
                                      const float* __restrict__ sc,
                                      const float* __restrict__ sh,
                                      const _Float16* __restrict__ lin,
                                      _Float16* __restrict__ lout,
                                      float* __restrict__ zout,
                                      int lane, int wm, int wn) {
    constexpr int NTW = N / 32;                 // 16-col tiles per wave
    constexpr int G   = (NTW % 4 == 0) ? 4 : 2; // tiles sharing one A fragment
    constexpr int KF  = (N / 16) * 512;         // halves between k-fragments
    const int lm = lane & 15;
    const int lh = lane >> 4;
    const _Float16* arow = lin + (size_t)(wm * 16 + lm) * STR;

    #pragma unroll 1
    for (int g = 0; g < NTW / G; ++g) {
        const int n0 = (wn * NTW + g * G) * 16;
        float8 acc[G];
        #pragma unroll
        for (int j = 0; j < G; ++j) acc[j] = (float8){};
        const _Float16* bw = wf + (size_t)(n0 / 16) * 512 + lane * 16;
        #pragma unroll 2
        for (int k0 = 0; k0 < K; k0 += 32) {
            AFrag a;
            // ISA 16-bit A layout: dwords 0..3 -> k = 8h+0..7 ; 4..7 -> k = 16+8h+0..7
            a.h[0] = *(const half8*)(arow + k0 + 8 * lh);
            a.h[1] = *(const half8*)(arow + k0 + 16 + 8 * lh);
            __builtin_prefetch((const void*)(bw + 2 * KF), 0, 1);
            #pragma unroll
            for (int j = 0; j < G; ++j) {
                half16 b = *(const half16*)(bw + j * 512);
                acc[j] = __builtin_amdgcn_wmma_f32_16x16x32_f16(
                    false, a.v, false, b, (short)0, acc[j], false, false);
            }
            bw += KF;
        }
        #pragma unroll
        for (int j = 0; j < G; ++j) {
            const int c = n0 + j * 16 + lm;     // this lane's output column
            if constexpr (BN) {
                const float bb_ = bias[c];
                const float s_  = sc[c];
                const float h_  = sh[c];
                #pragma unroll
                for (int r = 0; r < 8; ++r) {   // C/D layout: row = r + 8*lh
                    float hv = acc[j][r] + bb_;
                    hv = hv > 0.0f ? hv : 0.0f;
                    lout[(size_t)(wm * 16 + r + 8 * lh) * STR + c] = (_Float16)(hv * s_ + h_);
                }
            } else {
                const float bb_ = bias[c];
                #pragma unroll
                for (int r = 0; r < 8; ++r)
                    zout[(size_t)(wm * 16 + r + 8 * lh) * 64 + c] = acc[j][r] + bb_;
            }
        }
    }
}

// ---------------------------------------------------------------------------
// Kernel 3: fused 4-layer MLP. block = 256 thr = 8 waves; 64 rows, 1 tower.
// ---------------------------------------------------------------------------
__global__ void __launch_bounds__(256)
k_mlp(const _Float16* __restrict__ x,
      const _Float16* __restrict__ wf1, const _Float16* __restrict__ wf2,
      const _Float16* __restrict__ wf3, const _Float16* __restrict__ wf4,
      const float* __restrict__ b1, const float* __restrict__ sc1, const float* __restrict__ sh1,
      const float* __restrict__ b2, const float* __restrict__ sc2, const float* __restrict__ sh2,
      const float* __restrict__ b3, const float* __restrict__ sc3, const float* __restrict__ sh3,
      const float* __restrict__ b4,
      float* __restrict__ z) {
    extern __shared__ _Float16 smem[];
    _Float16* buf0 = smem;                  // 64 x STR halves
    _Float16* buf1 = smem + MT * STR;

    const int tid  = threadIdx.x;
    const int lane = tid & 31;
    const int wave = tid >> 5;
    const int wm   = wave >> 1;             // 4 x 16-row strips
    const int wn   = wave & 1;              // 2 x N/2-column strips
    const int t    = blockIdx.y;            // tower (0 = real head, 1 = imag head)
    const int row0 = blockIdx.x * MT;

    // Stage the 64x64 f16 input tile into LDS (dword-granular, coalesced)
    {
        const uint32_t* xs = (const uint32_t*)(x + ((size_t)t * ROWS + row0) * 64);
        for (int i = tid; i < MT * 32; i += 256) {   // 32 dwords per row
            const int r = i >> 5, c = i & 31;
            ((uint32_t*)(buf0 + (size_t)r * STR))[c] = xs[i];
        }
    }
    __syncthreads();

    layer<64, 1024, true >(wf1 + (size_t)t * 64 * 1024, b1 + t * 1024,
                           sc1 + t * 1024, sh1 + t * 1024,
                           buf0, buf1, nullptr, lane, wm, wn);
    __syncthreads();
    layer<1024, 512, true >(wf2 + (size_t)t * 1024 * 512, b2 + t * 512,
                           sc2 + t * 512, sh2 + t * 512,
                           buf1, buf0, nullptr, lane, wm, wn);
    __syncthreads();
    layer<512, 512, true >(wf3 + (size_t)t * 512 * 512, b3 + t * 512,
                           sc3 + t * 512, sh3 + t * 512,
                           buf0, buf1, nullptr, lane, wm, wn);
    __syncthreads();
    layer<512, 64, false>(wf4 + (size_t)t * 512 * 64, b4 + t * 64,
                          nullptr, nullptr,
                          buf1, nullptr, z + ((size_t)t * ROWS + row0) * 64,
                          lane, wm, wn);
}

// ---------------------------------------------------------------------------
// Kernel 4: heads + joint cross-tower L2 norm over TX + output transpose.
// out[((b*64 + u*32 + tx)*13 + sub)*2 + c], c=0 real tower, c=1 sigmoid tower.
// ---------------------------------------------------------------------------
__global__ void k_final(const float* __restrict__ z, float* __restrict__ out) {
    const int row = blockIdx.x;
    const int b   = row / SUB;
    const int sub = row - b * SUB;
    const int i   = threadIdx.x;            // col 0..63 : u = i&1, tx = i>>1
    const float z0 = z[(size_t)row * 64 + i];
    const float z1 = z[((size_t)ROWS + row) * 64 + i];
    const float zs = 1.0f / (1.0f + __expf(-z1));
    float s = z0 * z0 + zs * zs;
    // sum over same-parity lanes within this wave (parity == u)
    #pragma unroll
    for (int m = 2; m < 32; m <<= 1) s += __shfl_xor(s, m, 32);
    __shared__ float part[2][2];            // [wave][u]
    const int wv = i >> 5, u = i & 1;
    if ((i & 31) < 2) part[wv][i & 1] = s;  // lane0 -> even(u=0), lane1 -> odd(u=1)
    __syncthreads();
    const float rinv = rsqrtf(part[0][u] + part[1][u]);
    const size_t o = ((((size_t)b * 64) + u * 32 + (i >> 1)) * SUB + sub) * 2;
    out[o + 0] = z0 * rinv;
    out[o + 1] = zs * rinv;
}

// ---------------------------------------------------------------------------
// Host launcher
// ---------------------------------------------------------------------------
static inline size_t align256(size_t v) { return (v + 255) & ~(size_t)255; }

extern "C" void kernel_launch(void* const* d_in, const int* in_sizes, int n_in,
                              void* d_out, int out_size, void* d_ws, size_t ws_size,
                              hipStream_t stream) {
    (void)in_sizes; (void)n_in; (void)out_size; (void)ws_size;
    const float* V1  = (const float*)d_in[0];
    const float* V2  = (const float*)d_in[1];
    const float* W1  = (const float*)d_in[2];
    const float* b1  = (const float*)d_in[3];
    const float* g1  = (const float*)d_in[4];
    const float* be1 = (const float*)d_in[5];
    const float* m1  = (const float*)d_in[6];
    const float* v1  = (const float*)d_in[7];
    const float* W2  = (const float*)d_in[8];
    const float* b2  = (const float*)d_in[9];
    const float* g2  = (const float*)d_in[10];
    const float* be2 = (const float*)d_in[11];
    const float* m2  = (const float*)d_in[12];
    const float* v2  = (const float*)d_in[13];
    const float* W3  = (const float*)d_in[14];
    const float* b3  = (const float*)d_in[15];
    const float* g3  = (const float*)d_in[16];
    const float* be3 = (const float*)d_in[17];
    const float* m3  = (const float*)d_in[18];
    const float* v3  = (const float*)d_in[19];
    const float* W4  = (const float*)d_in[20];
    const float* b4  = (const float*)d_in[21];
    float* out = (float*)d_out;

    // Workspace carve-up (~45 MB total)
    char* ws = (char*)d_ws;
    size_t off = 0;
    _Float16* x   = (_Float16*)(ws + off); off = align256(off + (size_t)2 * ROWS * 64 * 2);
    _Float16* wf1 = (_Float16*)(ws + off); off = align256(off + (size_t)2 * 64 * 1024 * 2);
    _Float16* wf2 = (_Float16*)(ws + off); off = align256(off + (size_t)2 * 1024 * 512 * 2);
    _Float16* wf3 = (_Float16*)(ws + off); off = align256(off + (size_t)2 * 512 * 512 * 2);
    _Float16* wf4 = (_Float16*)(ws + off); off = align256(off + (size_t)2 * 512 * 64 * 2);
    float*    sc1 = (float*)   (ws + off); off = align256(off + (size_t)2 * 1024 * 4);
    float*    sh1 = (float*)   (ws + off); off = align256(off + (size_t)2 * 1024 * 4);
    float*    sc2 = (float*)   (ws + off); off = align256(off + (size_t)2 * 512 * 4);
    float*    sh2 = (float*)   (ws + off); off = align256(off + (size_t)2 * 512 * 4);
    float*    sc3 = (float*)   (ws + off); off = align256(off + (size_t)2 * 512 * 4);
    float*    sh3 = (float*)   (ws + off); off = align256(off + (size_t)2 * 512 * 4);
    float*    z   = (float*)   (ws + off); off = align256(off + (size_t)2 * ROWS * 64 * 4);

    // 1) input normalization -> f16 rows
    k_prep<<<ROWS, 64, 0, stream>>>(V1, V2, x);

    // 2) weight repack -> WMMA B-fragment layout; BN fold
    k_repack<64, 1024><<<(2 * 64 * 1024 + 255) / 256, 256, 0, stream>>>(W1, wf1);
    k_repack<1024, 512><<<(2 * 1024 * 512 + 255) / 256, 256, 0, stream>>>(W2, wf2);
    k_repack<512, 512><<<(2 * 512 * 512 + 255) / 256, 256, 0, stream>>>(W3, wf3);
    k_repack<512, 64><<<(2 * 512 * 64 + 255) / 256, 256, 0, stream>>>(W4, wf4);
    k_bnprep<1024><<<(2 * 1024 + 255) / 256, 256, 0, stream>>>(g1, be1, m1, v1, sc1, sh1);
    k_bnprep<512><<<(2 * 512 + 255) / 256, 256, 0, stream>>>(g2, be2, m2, v2, sc2, sh2);
    k_bnprep<512><<<(2 * 512 + 255) / 256, 256, 0, stream>>>(g3, be3, m3, v3, sc3, sh3);

    // 3) fused WMMA MLP (both towers via grid.y)
    const size_t shmem = (size_t)2 * MT * STR * sizeof(_Float16); // 264192 B
    dim3 grid(ROWS / MT, 2);
    k_mlp<<<grid, 256, shmem, stream>>>(x, wf1, wf2, wf3, wf4,
                                        b1, sc1, sh1,
                                        b2, sc2, sh2,
                                        b3, sc3, sh3,
                                        b4, z);

    // 4) heads + joint norm + transpose to (B, 64, SUB, 2)
    k_final<<<ROWS, 64, 0, stream>>>(z, out);
}
